// generator_33904471834671
// MI455X (gfx1250) — matrix operand order repaired and checked by reading, not compiled, not run on hardware
//
#include <hip/hip_runtime.h>
#include <hip/hip_bf16.h>

// ---------------------------------------------------------------------------
// Sequential 2-layer LSTM (H=1024) + MLP + argmax feedback, T=2048 steps.
// Persistent-kernel design for MI455X (gfx1250, wave32, WMMA):
//   * one-time repack of W_hh0/W_ih1/W_hh1 (f32 -> bf16 WMMA A-fragments)
//   * one-time transpose of W_ih0 (one-hot feedback => column gather)
//   * persistent kernel: 64 WGs x 256 thr, 3 grid barriers per step
//   * matvec via v_wmma_f32_16x16x32_bf16 with h broadcast into all B columns
// ---------------------------------------------------------------------------

#define IN_DIM 512
#define H_DIM  1024
#define OUT_DIM 512
#define T_STEPS 2048
#define R4H    4096          // 4*H
#define NWG    64
#define TPB    256

typedef __attribute__((ext_vector_type(16))) __bf16 v16bf;
typedef __attribute__((ext_vector_type(8)))  float  v8f;

union Frag16 {
    v16bf v;
    unsigned short u[16];
    unsigned int   d[8];
    uint4          q[2];
};

__device__ __host__ __forceinline__ unsigned short f2bf(float f) {
    union { float f; unsigned u; } x; x.f = f;
    unsigned r = x.u + 0x7FFFu + ((x.u >> 16) & 1u);   // round-to-nearest-even
    return (unsigned short)(r >> 16);
}

__device__ __forceinline__ float sigmoidf_(float x) { return 1.0f / (1.0f + expf(-x)); }

// One WMMA accumulate step: A = packed 16x32 bf16 weight tile (tm,kc),
// B = h[kc*32 .. kc*32+31] broadcast into all 16 columns.
__device__ __forceinline__ v8f wmma_mv(const unsigned short* __restrict__ pw,
                                       int tm, int kc,
                                       const unsigned short* __restrict__ hbf,
                                       int lane, v8f acc) {
    Frag16 a, b;
    const uint4* pa = (const uint4*)((const char*)pw +
                        (((size_t)(tm * 32 + kc)) << 10) + (size_t)lane * 32);
    a.q[0] = pa[0];
    a.q[1] = pa[1];
    // prefetch the next K-chunk of this tile row (8 chunks ahead = our stride)
    __builtin_prefetch((const char*)pa + 8 * 1024, 0, 1);

    const unsigned* hb = (const unsigned*)hbf;
    const int half = lane >> 4;
    const int base = kc * 16 + half * 8;     // dword index: (kc*32)/2 + half*8
#pragma unroll
    for (int j = 0; j < 8; ++j) b.d[j] = hb[base + j];

    return __builtin_amdgcn_wmma_f32_16x16x32_bf16(
        false, a.v, false, b.v, (short)0, acc, false, false);
}

struct Params {
    // inputs
    const float *Wih0, *state0, *bih0, *bhh0, *bih1, *bhh1;
    const float *Wp1, *bp1, *Wp2, *bp2, *Wp3, *bp3, *Wp4, *bp4;
    const float *h0in, *c0in;
    // workspace
    const unsigned short *pWhh0, *pWih1, *pWhh1;   // packed bf16 WMMA tiles
    const float *WihT;                              // W_ih0 transposed [IN][4H]
    float *h0f, *h1f;                               // [2][H] double-buffered
    float *c0f, *c1f;                               // [H]
    unsigned short *h0b, *h1b;                      // [2][H] bf16 copies
    int *idxp;
    unsigned *bar;
    float *out;                                     // [T]
};

__global__ void __launch_bounds__(TPB) init_kernel(unsigned* bar, int* idxp) {
    if (threadIdx.x == 0) { *bar = 0u; *idxp = 0; }
}

// Transpose W_ih0 (4096x512 row-major f32) -> WihT[k][row]
__global__ void __launch_bounds__(TPB) repack_wihT_kernel(const float* __restrict__ src,
                                                          float* __restrict__ dst) {
    unsigned i = blockIdx.x * blockDim.x + threadIdx.x;   // 0 .. 4096*512-1
    unsigned row = i >> 9, k = i & 511;
    dst[(size_t)k * R4H + row] = src[i];
}

// Pack a 4096x1024 f32 matrix into bf16 WMMA A-fragment tiles.
// Tile (tm,tk): 16 rows x 32 K. Lane l stores 16 bf16 in contiguous k order:
//   m = tm*16 + (l&15); half = l>>4;
//   pair p (VGPR p): klocal = 2p + 16*half
__global__ void __launch_bounds__(TPB) repack_wmma_kernel(
        const float* __restrict__ s0, const float* __restrict__ s1,
        const float* __restrict__ s2,
        unsigned short* __restrict__ d0, unsigned short* __restrict__ d1,
        unsigned short* __restrict__ d2) {
    const float* src; unsigned short* dst;
    if (blockIdx.y == 0)      { src = s0; dst = d0; }
    else if (blockIdx.y == 1) { src = s1; dst = d1; }
    else                      { src = s2; dst = d2; }

    unsigned gt   = blockIdx.x * blockDim.x + threadIdx.x;  // 0 .. 262143
    unsigned tile = gt >> 5, lane = gt & 31;
    unsigned tm = tile >> 5, tk = tile & 31;                // 256 x 32 tiles
    unsigned m = tm * 16 + (lane & 15), half = lane >> 4;
    const float* row = src + (size_t)m * H_DIM;

    union { unsigned short u[16]; uint4 q[2]; } tmp;
#pragma unroll
    for (int p = 0; p < 8; ++p) {
        int klocal = 2 * p + 16 * (int)half;
        int k = (int)tk * 32 + klocal;
        tmp.u[2 * p]     = f2bf(row[k]);
        tmp.u[2 * p + 1] = f2bf(row[k + 1]);
    }
    uint4* outp = (uint4*)((char*)dst + (size_t)tile * 1024 + (size_t)lane * 32);
    outp[0] = tmp.q[0];
    outp[1] = tmp.q[1];
}

__global__ void __launch_bounds__(TPB) lstm_persistent_kernel(Params p) {
    const int tid  = threadIdx.x;
    const int wg   = blockIdx.x;       // 0..63, owns hidden units [wg*16, wg*16+16)
    const int wave = tid >> 5;
    const int lane = tid & 31;
    const int u    = wg * 16;

    __shared__ float red[4][8][16];    // [gate][wave][unit] partial sums
    __shared__ float gsum[64];
    __shared__ float xc[64];
    __shared__ float z1[128], z2[192], z3[128];
    __shared__ float sv[TPB];
    __shared__ int   si[TPB];

    unsigned bar_target = 0;
    auto gbar = [&]() {
        __threadfence();
        __syncthreads();
        if (tid == 0) {
            bar_target += NWG;
            __hip_atomic_fetch_add(p.bar, 1u, __ATOMIC_RELEASE,
                                   __HIP_MEMORY_SCOPE_AGENT);
            while (__hip_atomic_load(p.bar, __ATOMIC_ACQUIRE,
                                     __HIP_MEMORY_SCOPE_AGENT) < bar_target) {
                __builtin_amdgcn_s_sleep(2);
            }
        }
        __syncthreads();
        __threadfence();
    };

    // ---- initialize recurrent state (parity-0 buffers) ----
    if (tid < 16) {
        float h0v = p.h0in[0 * H_DIM + u + tid];
        float h1v = p.h0in[1 * H_DIM + u + tid];
        p.h0f[u + tid] = h0v;  p.h0b[u + tid] = f2bf(h0v);
        p.h1f[u + tid] = h1v;  p.h1b[u + tid] = f2bf(h1v);
        p.c0f[u + tid] = p.c0in[0 * H_DIM + u + tid];
        p.c1f[u + tid] = p.c0in[1 * H_DIM + u + tid];
    }
    gbar();

    for (int t = 0; t < T_STEPS; ++t) {
        const int pp = t & 1, q = pp ^ 1;

        // ================= Phase A: LSTM layer 0 =================
        // x contribution: dense matvec at t==0, one-hot column gather after.
        if (t == 0) {
            if (tid < 64) {
                int g = tid >> 4, ml = tid & 15;
                int row = g * H_DIM + u + ml;
                const float* wr = p.Wih0 + (size_t)row * IN_DIM;
                float s = 0.f;
                for (int k = 0; k < IN_DIM; ++k) s += wr[k] * p.state0[k];
                xc[tid] = s;
            }
        } else {
            if (tid < 64) {
                int g = tid >> 4, ml = tid & 15;
                int row = g * H_DIM + u + ml;
                int idx = __hip_atomic_load(p.idxp, __ATOMIC_RELAXED,
                                            __HIP_MEMORY_SCOPE_AGENT);
                xc[tid] = p.WihT[(size_t)idx * R4H + row];
            }
        }

        const unsigned short* h0c = p.h0b + pp * H_DIM;
        for (int g = 0; g < 4; ++g) {
            v8f acc = {0.f, 0.f, 0.f, 0.f, 0.f, 0.f, 0.f, 0.f};
            const int tm = g * 64 + wg;
#pragma unroll
            for (int s = 0; s < 4; ++s) {
                int kc = wave + s * 8;
                acc = wmma_mv(p.pWhh0, tm, kc, h0c, lane, acc);
            }
            if ((lane & 15) == 0) {
                int half = lane >> 4;
#pragma unroll
                for (int j = 0; j < 8; ++j) red[g][wave][half * 8 + j] = acc[j];
            }
        }
        __syncthreads();
        if (tid < 64) {
            int g = tid >> 4, ml = tid & 15;
            int row = g * H_DIM + u + ml;
            float s = xc[tid] + p.bih0[row] + p.bhh0[row];
#pragma unroll
            for (int w = 0; w < 8; ++w) s += red[g][w][ml];
            gsum[tid] = s;
        }
        __syncthreads();
        if (tid < 16) {   // gate order: i, f, g, o
            float ig = gsum[tid], fg = gsum[16 + tid];
            float gg = gsum[32 + tid], og = gsum[48 + tid];
            float c  = p.c0f[u + tid];
            float cn = sigmoidf_(fg) * c + sigmoidf_(ig) * tanhf(gg);
            float hn = sigmoidf_(og) * tanhf(cn);
            p.c0f[u + tid] = cn;
            p.h0f[q * H_DIM + u + tid] = hn;
            p.h0b[q * H_DIM + u + tid] = f2bf(hn);
        }
        gbar();

        // ================= Phase B: LSTM layer 1 =================
        const unsigned short* h0n = p.h0b + q  * H_DIM;
        const unsigned short* h1c = p.h1b + pp * H_DIM;
        for (int g = 0; g < 4; ++g) {
            v8f acc = {0.f, 0.f, 0.f, 0.f, 0.f, 0.f, 0.f, 0.f};
            const int tm = g * 64 + wg;
#pragma unroll
            for (int s = 0; s < 4; ++s) {
                int kc = wave + s * 8;
                acc = wmma_mv(p.pWih1, tm, kc, h0n, lane, acc);
                acc = wmma_mv(p.pWhh1, tm, kc, h1c, lane, acc);
            }
            if ((lane & 15) == 0) {
                int half = lane >> 4;
#pragma unroll
                for (int j = 0; j < 8; ++j) red[g][wave][half * 8 + j] = acc[j];
            }
        }
        __syncthreads();
        if (tid < 64) {
            int g = tid >> 4, ml = tid & 15;
            int row = g * H_DIM + u + ml;
            float s = p.bih1[row] + p.bhh1[row];
#pragma unroll
            for (int w = 0; w < 8; ++w) s += red[g][w][ml];
            gsum[tid] = s;
        }
        __syncthreads();
        if (tid < 16) {
            float ig = gsum[tid], fg = gsum[16 + tid];
            float gg = gsum[32 + tid], og = gsum[48 + tid];
            float c  = p.c1f[u + tid];
            float cn = sigmoidf_(fg) * c + sigmoidf_(ig) * tanhf(gg);
            float hn = sigmoidf_(og) * tanhf(cn);
            p.c1f[u + tid] = cn;
            p.h1f[q * H_DIM + u + tid] = hn;
            p.h1b[q * H_DIM + u + tid] = f2bf(hn);
        }
        gbar();

        // ================= Phase C: MLP + argmax (WG0 only) =================
        if (wg == 0) {
            const float* h1n = p.h1f + q * H_DIM;
            if (tid < 128) {
                const float* wr = p.Wp1 + (size_t)tid * H_DIM;
                float s = p.bp1[tid];
                for (int k = 0; k < H_DIM; ++k) s += wr[k] * h1n[k];
                z1[tid] = fmaxf(s, 0.f);
            }
            __syncthreads();
            if (tid < 192) {
                const float* wr = p.Wp2 + (size_t)tid * 128;
                float s = p.bp2[tid];
                for (int k = 0; k < 128; ++k) s += wr[k] * z1[k];
                z2[tid] = fmaxf(s, 0.f);
            }
            __syncthreads();
            if (tid < 128) {
                const float* wr = p.Wp3 + (size_t)tid * 192;
                float s = p.bp3[tid];
                for (int k = 0; k < 192; ++k) s += wr[k] * z2[k];
                z3[tid] = fmaxf(s, 0.f);
            }
            __syncthreads();
            {   // logits (softmax is monotonic -> argmax of logits)
                int r0 = tid, r1 = tid + 256;
                const float* w0 = p.Wp4 + (size_t)r0 * 128;
                const float* w1 = p.Wp4 + (size_t)r1 * 128;
                float v0 = p.bp4[r0], v1 = p.bp4[r1];
                for (int k = 0; k < 128; ++k) {
                    v0 += w0[k] * z3[k];
                    v1 += w1[k] * z3[k];
                }
                float bv = v0; int bi = r0;
                if (v1 > bv) { bv = v1; bi = r1; }
                sv[tid] = bv; si[tid] = bi;
            }
            __syncthreads();
            for (int s = 128; s > 0; s >>= 1) {
                if (tid < s) {
                    float ov = sv[tid + s]; int oi = si[tid + s];
                    if (ov > sv[tid] || (ov == sv[tid] && oi < si[tid])) {
                        sv[tid] = ov; si[tid] = oi;   // first-max tie-break
                    }
                }
                __syncthreads();
            }
            if (tid == 0) {
                *p.idxp = si[0];
                p.out[t] = (float)si[0];
            }
        }
        gbar();
    }
}

extern "C" void kernel_launch(void* const* d_in, const int* in_sizes, int n_in,
                              void* d_out, int out_size, void* d_ws, size_t ws_size,
                              hipStream_t stream) {
    (void)in_sizes; (void)n_in; (void)out_size; (void)ws_size;

    const float* h0in   = (const float*)d_in[1];
    const float* c0in   = (const float*)d_in[2];
    const float* state0 = (const float*)d_in[3];
    const float* Wih0   = (const float*)d_in[4];
    const float* Whh0   = (const float*)d_in[5];
    const float* bih0   = (const float*)d_in[6];
    const float* bhh0   = (const float*)d_in[7];
    const float* Wih1   = (const float*)d_in[8];
    const float* Whh1   = (const float*)d_in[9];
    const float* bih1   = (const float*)d_in[10];
    const float* bhh1   = (const float*)d_in[11];
    const float* Wp1    = (const float*)d_in[12];
    const float* bp1    = (const float*)d_in[13];
    const float* Wp2    = (const float*)d_in[14];
    const float* bp2    = (const float*)d_in[15];
    const float* Wp3    = (const float*)d_in[16];
    const float* bp3    = (const float*)d_in[17];
    const float* Wp4    = (const float*)d_in[18];
    const float* bp4    = (const float*)d_in[19];

    // ---- workspace carve-up ----
    char* ws = (char*)d_ws;
    size_t off = 0;
    auto take = [&](size_t bytes) -> void* {
        void* r = ws + off;
        off = (off + bytes + 255) & ~(size_t)255;
        return r;
    };
    unsigned short* pWhh0 = (unsigned short*)take((size_t)R4H * H_DIM * 2);
    unsigned short* pWih1 = (unsigned short*)take((size_t)R4H * H_DIM * 2);
    unsigned short* pWhh1 = (unsigned short*)take((size_t)R4H * H_DIM * 2);
    float* WihT           = (float*)take((size_t)IN_DIM * R4H * 4);
    float* h0f            = (float*)take(2 * H_DIM * 4);
    float* h1f            = (float*)take(2 * H_DIM * 4);
    float* c0f            = (float*)take(H_DIM * 4);
    float* c1f            = (float*)take(H_DIM * 4);
    unsigned short* h0b   = (unsigned short*)take(2 * H_DIM * 2);
    unsigned short* h1b   = (unsigned short*)take(2 * H_DIM * 2);
    int* idxp             = (int*)take(256);
    unsigned* bar         = (unsigned*)take(256);

    // 1) reset barrier / idx (d_ws may be poisoned)
    init_kernel<<<1, TPB, 0, stream>>>(bar, idxp);
    // 2) pack the three 4096x1024 matrices into bf16 WMMA A-fragments
    repack_wmma_kernel<<<dim3(1024, 3), TPB, 0, stream>>>(
        Whh0, Wih1, Whh1, pWhh0, pWih1, pWhh1);
    // 3) transpose W_ih0 for one-hot column gather
    repack_wihT_kernel<<<(R4H * IN_DIM) / TPB, TPB, 0, stream>>>(Wih0, WihT);

    // 4) persistent sequential loop
    Params p;
    p.Wih0 = Wih0; p.state0 = state0;
    p.bih0 = bih0; p.bhh0 = bhh0; p.bih1 = bih1; p.bhh1 = bhh1;
    p.Wp1 = Wp1; p.bp1 = bp1; p.Wp2 = Wp2; p.bp2 = bp2;
    p.Wp3 = Wp3; p.bp3 = bp3; p.Wp4 = Wp4; p.bp4 = bp4;
    p.h0in = h0in; p.c0in = c0in;
    p.pWhh0 = pWhh0; p.pWih1 = pWih1; p.pWhh1 = pWhh1;
    p.WihT = WihT;
    p.h0f = h0f; p.h1f = h1f; p.c0f = c0f; p.c1f = c1f;
    p.h0b = h0b; p.h1b = h1b;
    p.idxp = idxp; p.bar = bar;
    p.out = (float*)d_out;

    lstm_persistent_kernel<<<NWG, TPB, 0, stream>>>(p);
}
